// SelfAttention_3865470566964
// MI455X (gfx1250) — matrix operand order repaired and checked by reading, not compiled, not run on hardware
//
#include <hip/hip_runtime.h>
#include <hip/hip_bf16.h>
#include <stdint.h>

#define T_SEQ 4096
#define EMBED 1024
#define HEADS 16
#define HDIM  64

typedef __bf16 bf16_t;
typedef __attribute__((ext_vector_type(16))) __bf16 v16bf;
typedef __attribute__((ext_vector_type(8)))  __bf16 v8bf;
typedef __attribute__((ext_vector_type(8)))  float  v8f;
typedef __attribute__((ext_vector_type(4)))  unsigned int u32x4;
typedef __attribute__((ext_vector_type(8)))  int i32x8;
typedef __attribute__((ext_vector_type(4)))  int i32x4;

#if __has_builtin(__builtin_amdgcn_tensor_load_to_lds) && __has_builtin(__builtin_amdgcn_s_wait_tensorcnt)
#define ATTN_USE_TDM 1
#else
#define ATTN_USE_TDM 0
#endif

__device__ __forceinline__ v8f wmma_bf16(v16bf a, v16bf b, v8f c) {
    // D = A(16x32) * B(32x16) + C, f32 accumulate
    return __builtin_amdgcn_wmma_f32_16x16x32_bf16(
        false, a, false, b, (short)0, c, false, false);
}

// A-operand (16x32 bf16): lane(lo,hi) holds row `lo`; two 16B chunks at
// K = 8*hi and K = 16+8*hi.
__device__ __forceinline__ v16bf load_a(const bf16_t* row, int hi) {
    union { v16bf v; v8bf h[2]; } u;
    u.h[0] = *reinterpret_cast<const v8bf*>(row + 8 * hi);
    u.h[1] = *reinterpret_cast<const v8bf*>(row + 16 + 8 * hi);
    return u.v;
}

// B-operand (32x16 bf16): lane(lo,hi) holds column `lo`; two 16B chunks at
// K = 8*hi and K = 16+8*hi (matrix stored column-major = row of transposed matrix).
__device__ __forceinline__ v16bf load_b(const bf16_t* col, int hi) {
    union { v16bf v; v8bf h[2]; } u;
    u.h[0] = *reinterpret_cast<const v8bf*>(col + 8 * hi);
    u.h[1] = *reinterpret_cast<const v8bf*>(col + 16 + 8 * hi);
    return u.v;
}

#if ATTN_USE_TDM
// Build a TDM D# (groups 0/1 per CDNA5 ISA sect.8) and issue
// TENSOR_LOAD_TO_LDS (6-arg clang-23 form; unused groups zero-filled).
// data_size = 4B units; tile = tile_d0_dw x tile_d1 rows with row stride
// stride0_dw; rows packed contiguously at lds_byte_off.
__device__ __forceinline__ void tdm_load(unsigned lds_byte_off, const void* gptr,
                                         unsigned tile_d0_dw, unsigned tile_d1,
                                         unsigned stride0_dw) {
    const unsigned long long ga = (unsigned long long)(uintptr_t)gptr;
    u32x4 g0;
    g0[0] = 1u;                                    // count=1, user descriptor
    g0[1] = lds_byte_off;                          // lds_addr
    g0[2] = (unsigned)ga;                          // global_addr[31:0]
    g0[3] = (unsigned)((ga >> 32) & 0x01FFFFFFu)   // global_addr[56:32]
          | 0x80000000u;                           // type=2 ("image")
    const unsigned td = 1u << 20;                  // big tensor dims: no OOB clip
    i32x8 g1;
    g1[0] = (int)(2u << 16);                       // wg_mask=0, data_size=4B
    g1[1] = (int)((td & 0xFFFFu) << 16);                              // td0[15:0]
    g1[2] = (int)(((td >> 16) & 0xFFFFu) | ((td & 0xFFFFu) << 16));   // td0[31:16]|td1[15:0]
    g1[3] = (int)(((td >> 16) & 0xFFFFu) | ((tile_d0_dw & 0xFFFFu) << 16)); // td1[31:16]|tile_dim0
    g1[4] = (int)(tile_d1 & 0xFFFFu);              // tile_dim1 (0 => single row), tile_dim2=0
    g1[5] = (int)stride0_dw;                       // tensor_dim0_stride[31:0]
    g1[6] = 0;
    g1[7] = 0;
    const i32x4 z4 = {0, 0, 0, 0};                 // groups 2/3: unused (<=2D)
    const i32x8 z8 = {0, 0, 0, 0, 0, 0, 0, 0};
    __builtin_amdgcn_tensor_load_to_lds(g0, g1, z4, z4, z8, 0);
}
__device__ __forceinline__ unsigned lds_off(const void* p) {
    return (unsigned)(uintptr_t)p;                 // generic LDS addr low 32 bits
}
#endif

// ---------------------------------------------------------------- converts
__global__ void cvt_kernel(const float* __restrict__ in, bf16_t* __restrict__ out, int n) {
    int i = blockIdx.x * blockDim.x + threadIdx.x;
    int stride = gridDim.x * blockDim.x;
    for (; i < n; i += stride) out[i] = (bf16_t)in[i];
}

// in[rows][cols] (f32) -> out[cols][rows] (bf16)
__global__ void transpose_cvt_kernel(const float* __restrict__ in, bf16_t* __restrict__ out,
                                     int rows, int cols) {
    int i = blockIdx.x * blockDim.x + threadIdx.x;
    int n = rows * cols;
    int stride = gridDim.x * blockDim.x;
    for (; i < n; i += stride) {
        int r = i / cols, c = i - r * cols;
        out[c * rows + r] = (bf16_t)in[i];
    }
}

// ---------------------------------------------------------- QKV projection
// 64x64 output tile per wave: 4 A-frags x 4 B-frags -> 16 independent WMMAs
// per k-step (32 FLOP/B from L2).  Q scaled by 1/8; V written transposed.
__global__ void __launch_bounds__(256) qkv_gemm_kernel(
        const bf16_t* __restrict__ xb,   // [T][C]
        const bf16_t* __restrict__ Wt,   // [3C][C] (W^T)
        const float*  __restrict__ bias, // [3C]
        bf16_t* __restrict__ Q,          // [H][T][64]
        bf16_t* __restrict__ K,          // [H][T][64]
        bf16_t* __restrict__ Vt) {       // [H][64][T]
    const int gtid = blockIdx.x * blockDim.x + threadIdx.x;
    const int wave = gtid >> 5;
    const int lane = threadIdx.x & 31;
    const int lo = lane & 15, hi = lane >> 4;
    const int NT = (3 * EMBED) / 64;        // 48
    const int tm = wave / NT;
    const int tn = wave - tm * NT;

    const bf16_t* arow0 = xb + (size_t)(tm * 64 + lo) * EMBED;
    const bf16_t* brow0 = Wt + (size_t)(tn * 64 + lo) * EMBED;

    v8f acc[4][4] = {};
    for (int k0 = 0; k0 < EMBED; k0 += 32) {
        __builtin_prefetch(arow0 + k0 + 128, 0, 1);
        __builtin_prefetch(brow0 + k0 + 128, 0, 1);
        v16bf a[4], b[4];
#pragma unroll
        for (int i = 0; i < 4; ++i) a[i] = load_a(arow0 + (size_t)(16 * i) * EMBED + k0, hi);
#pragma unroll
        for (int j = 0; j < 4; ++j) b[j] = load_b(brow0 + (size_t)(16 * j) * EMBED + k0, hi);
#pragma unroll
        for (int i = 0; i < 4; ++i)
#pragma unroll
            for (int j = 0; j < 4; ++j)
                acc[i][j] = wmma_bf16(a[i], b[j], acc[i][j]);
    }

#pragma unroll
    for (int j = 0; j < 4; ++j) {
        const int jc = tn * 64 + j * 16 + lo;
        const float bj = bias[jc];
        const int s  = jc >> 10;                // 0=Q 1=K 2=V
        const int hd = jc & 1023;
        const int h  = hd >> 6;
        const int d  = hd & 63;
#pragma unroll
        for (int i = 0; i < 4; ++i) {
#pragma unroll
            for (int v = 0; v < 8; ++v) {
                const int t = tm * 64 + i * 16 + v + 8 * hi;
                const float val = acc[i][j][v] + bj;
                if (s == 0)      Q[((size_t)h * T_SEQ + t) * HDIM + d] = (bf16_t)(val * 0.125f);
                else if (s == 1) K[((size_t)h * T_SEQ + t) * HDIM + d] = (bf16_t)val;
                else             Vt[((size_t)h * HDIM + d) * T_SEQ + t] = (bf16_t)val;
            }
        }
    }
}

// ------------------------------------------------------- flash attention
// Workgroup = 8 waves sharing (head, 256-query stripe).  Each 32-key block's
// K (4KB contiguous) and V (64 strided rows, a true 2-D tile) is staged ONCE
// per workgroup into double-buffered LDS — by the Tensor Data Mover when
// available — then all 8 waves consume it (8x cut in L2 traffic).
// S^T = K.Q^T so softmaxed P lands directly in the WMMA A-operand layout.
__global__ void __launch_bounds__(256) attention_kernel(
        const bf16_t* __restrict__ Q,   // [H][T][64]
        const bf16_t* __restrict__ K,   // [H][T][64]
        const bf16_t* __restrict__ Vt,  // [H][64][T]
        bf16_t* __restrict__ O) {       // [T][C], col = h*64+d
    __shared__ alignas(64) bf16_t ldsK[2][32 * HDIM];   // 32 keys x 64 d
    __shared__ alignas(64) bf16_t ldsV[2][HDIM * 32];   // 64 d   x 32 keys

    const int tid  = threadIdx.x;
    const int lane = tid & 31;
    const int wv   = tid >> 5;          // wave in block: 0..7
    const int lo = lane & 15, hi = lane >> 4;
    const int h      = blockIdx.x >> 4;
    const int stripe = blockIdx.x & 15;
    const int qs     = stripe * 256;
    const int qbase  = qs + 32 * wv;    // this wave's 32 queries

    const bf16_t* Qh = Q  + (size_t)h * T_SEQ * HDIM;
    const bf16_t* Kh = K  + (size_t)h * T_SEQ * HDIM;
    const bf16_t* Vh = Vt + (size_t)h * HDIM * T_SEQ;

    // Q^T B-operand fragments for both query sub-tiles
    v16bf qf[2][2];
#pragma unroll
    for (int u = 0; u < 2; ++u) {
        const bf16_t* qrow = Qh + (size_t)(qbase + 16 * u + lo) * HDIM;
        qf[u][0] = load_b(qrow, hi);
        qf[u][1] = load_b(qrow + 32, hi);
    }

    float m[2] = {-3.0e38f, -3.0e38f};
    float l[2] = {0.0f, 0.0f};
    v8f o[2][4] = {};

    auto stage = [&](int buf, int kb) {
#if ATTN_USE_TDM
        if (tid < 32) {   // one TDM issue per workgroup (wave 0)
            tdm_load(lds_off(&ldsK[buf][0]), Kh + (size_t)kb * HDIM,
                     /*tile_d0=*/ (32 * HDIM * 2) / 4, /*tile_d1=*/ 0, /*stride=*/ 1024);
            tdm_load(lds_off(&ldsV[buf][0]), Vh + kb,
                     /*tile_d0=*/ (32 * 2) / 4, /*tile_d1=*/ HDIM,
                     /*stride=*/ (T_SEQ * 2) / 4);
        }
#else
        // cooperative fallback: 256 threads x 32 B
        if (tid < 128) {
            const v8bf* src = reinterpret_cast<const v8bf*>(Kh + (size_t)kb * HDIM) + tid * 2;
            v8bf* dst = reinterpret_cast<v8bf*>(&ldsK[buf][0]) + tid * 2;
            dst[0] = src[0]; dst[1] = src[1];
        } else {
            const int i = tid - 128, d = i >> 1, half = i & 1;
            const v8bf* src =
                reinterpret_cast<const v8bf*>(Vh + (size_t)d * T_SEQ + kb + half * 16);
            v8bf* dst = reinterpret_cast<v8bf*>(&ldsV[buf][d * 32 + half * 16]);
            dst[0] = src[0]; dst[1] = src[1];
        }
#endif
    };

    const int nb = 8 * (stripe + 1);    // key blocks: keys [0, qs+256)
    stage(0, 0);
    for (int i = 0; i < nb; ++i) {
        const int kb  = i * 32;
        const int buf = i & 1;
#if ATTN_USE_TDM
        if (tid < 32) __builtin_amdgcn_s_wait_tensorcnt(0);
#endif
        __syncthreads();                 // block i resident in LDS[buf]
        if (i + 1 < nb) stage(buf ^ 1, kb + 32);   // overlap next block's DMA

        if (kb < qbase + 32) {           // causal: this wave still needs keys
            // K rows as A-operand from LDS
            v16bf kf[2][2];
#pragma unroll
            for (int kt = 0; kt < 2; ++kt) {
                const bf16_t* krow = &ldsK[buf][(16 * kt + lo) * HDIM];
                kf[kt][0] = load_a(krow, hi);
                kf[kt][1] = load_a(krow + 32, hi);
            }
            // V columns as B-operand from LDS
            v16bf vf[4];
#pragma unroll
            for (int dt = 0; dt < 4; ++dt)
                vf[dt] = load_b(&ldsV[buf][(16 * dt + lo) * 32], hi);

#pragma unroll
            for (int u = 0; u < 2; ++u) {
                v8f st0 = {}, st1 = {};
                st0 = wmma_bf16(kf[0][0], qf[u][0], st0);
                st0 = wmma_bf16(kf[0][1], qf[u][1], st0);
                st1 = wmma_bf16(kf[1][0], qf[u][0], st1);
                st1 = wmma_bf16(kf[1][1], qf[u][1], st1);

                const int q = qbase + 16 * u + lo;
                if (kb + 31 > qbase + 16 * u) {   // diagonal blocks
#pragma unroll
                    for (int v = 0; v < 8; ++v) {
                        if (kb + v + 8 * hi > q)      st0[v] = -3.0e38f;
                        if (kb + 16 + v + 8 * hi > q) st1[v] = -3.0e38f;
                    }
                }

                float mloc = -3.0e38f;
#pragma unroll
                for (int v = 0; v < 8; ++v) {
                    mloc = fmaxf(mloc, st0[v]);
                    mloc = fmaxf(mloc, st1[v]);
                }
                mloc = fmaxf(mloc, __shfl_xor(mloc, 16));
                const float mnew  = fmaxf(m[u], mloc);
                const float alpha = __expf(m[u] - mnew);

                float sl = 0.0f;
                v16bf pa;                  // P directly in A-operand layout
#pragma unroll
                for (int v = 0; v < 8; ++v) {
                    const float p0 = __expf(st0[v] - mnew);
                    const float p1 = __expf(st1[v] - mnew);
                    sl += p0 + p1;
                    pa[v]     = (__bf16)p0;
                    pa[8 + v] = (__bf16)p1;
                }
                sl += __shfl_xor(sl, 16);
                l[u] = l[u] * alpha + sl;
                m[u] = mnew;

#pragma unroll
                for (int v = 0; v < 8; ++v) {
                    const float ar = __shfl(alpha, v + 8 * hi);
                    o[u][0][v] *= ar; o[u][1][v] *= ar;
                    o[u][2][v] *= ar; o[u][3][v] *= ar;
                }
#pragma unroll
                for (int dt = 0; dt < 4; ++dt)
                    o[u][dt] = wmma_bf16(pa, vf[dt], o[u][dt]);
            }
        }
        __syncthreads();                 // before LDS[buf] is overwritten
    }

#pragma unroll
    for (int u = 0; u < 2; ++u) {
#pragma unroll
        for (int v = 0; v < 8; ++v) {
            const float linv = 1.0f / __shfl(l[u], v + 8 * hi);
            const int t = qbase + 16 * u + v + 8 * hi;
            bf16_t* orow = O + (size_t)t * EMBED + h * HDIM + lo;
            orow[0]  = (bf16_t)(o[u][0][v] * linv);
            orow[16] = (bf16_t)(o[u][1][v] * linv);
            orow[32] = (bf16_t)(o[u][2][v] * linv);
            orow[48] = (bf16_t)(o[u][3][v] * linv);
        }
    }
}

// --------------------------------------------------------- out projection
__global__ void __launch_bounds__(256) out_gemm_kernel(
        const bf16_t* __restrict__ Ab,  // [T][C]
        const bf16_t* __restrict__ Wt,  // [C][C] (W^T)
        const float*  __restrict__ bias,// [C]
        float* __restrict__ out) {      // [T][C] f32
    const int gtid = blockIdx.x * blockDim.x + threadIdx.x;
    const int wave = gtid >> 5;
    const int lane = threadIdx.x & 31;
    const int lo = lane & 15, hi = lane >> 4;
    const int NT = EMBED / 64;              // 16
    const int tm = wave / NT;
    const int tn = wave - tm * NT;

    const bf16_t* arow0 = Ab + (size_t)(tm * 64 + lo) * EMBED;
    const bf16_t* brow0 = Wt + (size_t)(tn * 64 + lo) * EMBED;

    v8f acc[4][4] = {};
    for (int k0 = 0; k0 < EMBED; k0 += 32) {
        __builtin_prefetch(arow0 + k0 + 128, 0, 1);
        __builtin_prefetch(brow0 + k0 + 128, 0, 1);
        v16bf a[4], b[4];
#pragma unroll
        for (int i = 0; i < 4; ++i) a[i] = load_a(arow0 + (size_t)(16 * i) * EMBED + k0, hi);
#pragma unroll
        for (int j = 0; j < 4; ++j) b[j] = load_b(brow0 + (size_t)(16 * j) * EMBED + k0, hi);
#pragma unroll
        for (int i = 0; i < 4; ++i)
#pragma unroll
            for (int j = 0; j < 4; ++j)
                acc[i][j] = wmma_bf16(a[i], b[j], acc[i][j]);
    }

#pragma unroll
    for (int j = 0; j < 4; ++j) {
        const int jc = tn * 64 + j * 16 + lo;
        const float bj = bias[jc];
#pragma unroll
        for (int i = 0; i < 4; ++i)
#pragma unroll
            for (int v = 0; v < 8; ++v) {
                const int t = tm * 64 + i * 16 + v + 8 * hi;
                out[(size_t)t * EMBED + jc] = acc[i][j][v] + bj;
            }
    }
}

// ------------------------------------------------------------------ launch
extern "C" void kernel_launch(void* const* d_in, const int* in_sizes, int n_in,
                              void* d_out, int out_size, void* d_ws, size_t ws_size,
                              hipStream_t stream) {
    const float* x     = (const float*)d_in[0];
    const float* W_qkv = (const float*)d_in[1];
    const float* b_qkv = (const float*)d_in[2];
    const float* W_out = (const float*)d_in[3];
    const float* b_out = (const float*)d_in[4];
    float* out = (float*)d_out;

    char* ws = (char*)d_ws;
    size_t off = 0;
    auto alloc = [&](size_t bytes) -> char* {
        char* p = ws + off;
        off += (bytes + 255) & ~(size_t)255;
        return p;
    };
    bf16_t* xb    = (bf16_t*)alloc((size_t)T_SEQ * EMBED * 2);
    bf16_t* Wqkvt = (bf16_t*)alloc((size_t)3 * EMBED * EMBED * 2);
    bf16_t* Woutt = (bf16_t*)alloc((size_t)EMBED * EMBED * 2);
    bf16_t* Qb    = (bf16_t*)alloc((size_t)HEADS * T_SEQ * HDIM * 2);
    bf16_t* Kb    = (bf16_t*)alloc((size_t)HEADS * T_SEQ * HDIM * 2);
    bf16_t* Vtb   = (bf16_t*)alloc((size_t)HEADS * HDIM * T_SEQ * 2);
    bf16_t* Ob    = (bf16_t*)alloc((size_t)T_SEQ * EMBED * 2);

    cvt_kernel<<<1024, 256, 0, stream>>>(x, xb, T_SEQ * EMBED);
    transpose_cvt_kernel<<<1024, 256, 0, stream>>>(W_qkv, Wqkvt, EMBED, 3 * EMBED);
    transpose_cvt_kernel<<<1024, 256, 0, stream>>>(W_out, Woutt, EMBED, EMBED);

    qkv_gemm_kernel<<<(64 * 48) / 8, 256, 0, stream>>>(xb, Wqkvt, b_qkv, Qb, Kb, Vtb);

    // 16 heads x 16 query stripes; 8 waves per block share LDS-staged K/V
    attention_kernel<<<HEADS * 16, 256, 0, stream>>>(Qb, Kb, Vtb, Ob);

    out_gemm_kernel<<<(64 * 16) / 8, 256, 0, stream>>>(Ob, Woutt, b_out, out);
}
